// EncoderDecoderRNN_59210419142755
// MI455X (gfx1250) — matrix operand order repaired and checked by reading, not compiled, not run on hardware
//
#include <hip/hip_runtime.h>
#include <hip/hip_bf16.h>

// ---------------------------------------------------------------------------
// Problem constants (from reference): B=32, S=T=128, V=32000, E=H=256
// ---------------------------------------------------------------------------
#define BB   32
#define SS   128
#define TT   128
#define VV   32000
#define HH   256
#define NEGF (-1.0e9f)

// W_hh LDS staging: 256 rows padded to 258 floats (1032 B rows -> per-lane
// ds_load_b64 across lanes lands on distinct bank pairs; 2j mod 64 distinct
// for j in 0..31). Total 264,192 B < 320 KB WGP LDS.
#define WHH_PAD      258
#define WHH_LDS_BYTES ((size_t)HH * WHH_PAD * 4)

typedef __bf16 bf16_t;
typedef __attribute__((ext_vector_type(16))) __bf16 v16bf;
typedef __attribute__((ext_vector_type(8)))  __bf16 v8bf;
typedef __attribute__((ext_vector_type(8)))  float  v8f;

static __device__ __forceinline__ bf16_t f2bf(float f) {
    union { float f; unsigned u; } x; x.f = f;
    unsigned r = x.u + 0x7fffu + ((x.u >> 16) & 1u);   // round-to-nearest-even
    union { unsigned short s; bf16_t b; } y; y.s = (unsigned short)(r >> 16);
    return y.b;
}

// ---------------------------------------------------------------------------
// Elementwise f32 -> bf16 convert (weights)
// ---------------------------------------------------------------------------
__global__ void f32_to_bf16_kernel(const float* __restrict__ src,
                                   bf16_t* __restrict__ dst, size_t n) {
    size_t i = (size_t)blockIdx.x * blockDim.x + threadIdx.x;
    size_t stride = (size_t)gridDim.x * blockDim.x;
    for (; i < n; i += stride) dst[i] = f2bf(src[i]);
}

// ---------------------------------------------------------------------------
// Embedding gather + convert: X[r, e] = bf16(emb[idx[r], e]);  one row / block
// ---------------------------------------------------------------------------
__global__ void gather_embed_kernel(const float* __restrict__ emb,
                                    const int* __restrict__ idx,
                                    bf16_t* __restrict__ X) {
    int r = blockIdx.x;            // 0 .. B*S-1
    int e = threadIdx.x;           // 0 .. 255
    int id = idx[r];
    X[(size_t)r * HH + e] = f2bf(emb[(size_t)id * HH + e]);
}

// ---------------------------------------------------------------------------
// bf16 WMMA GEMM:  C[M,N] = A[M,K] * B[N,K]^T (+ bias[N])
// Each wave computes a 64(M) x 16(N) strip: 4 M-tiles sharing one B fragment
// per k-step -> 4 independent accumulator chains (hides WMMA RAW latency) and
// 4x less B (weight) traffic from L2. K == 256. M % 64 == 0, N % 16 == 0.
// NT=true uses non-temporal stores (output >> L2; keep W_out resident).
// ---------------------------------------------------------------------------
template <bool NT>
__global__ void gemm_bf16_wmma_kernel(const bf16_t* __restrict__ A,
                                      const bf16_t* __restrict__ Bw,
                                      const float* __restrict__ bias,
                                      float* __restrict__ C,
                                      int M, int N, int K, int totalGroups) {
    const int wave = threadIdx.x >> 5;
    const int lane = threadIdx.x & 31;
    const int g = blockIdx.x * (blockDim.x >> 5) + wave;
    if (g >= totalGroups) return;                 // wave-uniform exit

    const int ntN = N >> 4;
    const int mg = g / ntN;                       // 64-row group
    const int nt = g - mg * ntN;
    const int r0 = mg << 6;
    const int n0 = nt << 4;

    const int lhalf = lane >> 4;                  // 0: lanes 0-15, 1: lanes 16-31
    const int lmod  = lane & 15;

    // A: lane holds row; contiguous K halves (0-15 vs 16-31)
    const bf16_t* arow = A  + (size_t)(r0 + lmod) * K + lhalf * 16;
    // B: lane holds column; contiguous K halves (0-15 vs 16-31)
    const bf16_t* brow = Bw + (size_t)(n0 + lmod) * K + lhalf * 16;
    const size_t astride = (size_t)16 * K;        // next M-tile

    v8f acc0 = {}, acc1 = {}, acc2 = {}, acc3 = {};

#pragma unroll
    for (int kb = 0; kb < 256; kb += 32) {
        v16bf bfrag = *(const v16bf*)(brow + kb);

        v8bf lo0 = *(const v8bf*)(arow + kb);
        v8bf hi0 = *(const v8bf*)(arow + kb + 8);
        v8bf lo1 = *(const v8bf*)(arow + astride + kb);
        v8bf hi1 = *(const v8bf*)(arow + astride + kb + 8);
        v8bf lo2 = *(const v8bf*)(arow + 2 * astride + kb);
        v8bf hi2 = *(const v8bf*)(arow + 2 * astride + kb + 8);
        v8bf lo3 = *(const v8bf*)(arow + 3 * astride + kb);
        v8bf hi3 = *(const v8bf*)(arow + 3 * astride + kb + 8);

        v16bf a0, a1, a2, a3;
#pragma unroll
        for (int i = 0; i < 8; ++i) {
            a0[i] = lo0[i]; a0[i + 8] = hi0[i];
            a1[i] = lo1[i]; a1[i + 8] = hi1[i];
            a2[i] = lo2[i]; a2[i + 8] = hi2[i];
            a3[i] = lo3[i]; a3[i + 8] = hi3[i];
        }
        acc0 = __builtin_amdgcn_wmma_f32_16x16x32_bf16(false, a0, false, bfrag,
                                                       (short)0, acc0, false, false);
        acc1 = __builtin_amdgcn_wmma_f32_16x16x32_bf16(false, a1, false, bfrag,
                                                       (short)0, acc1, false, false);
        acc2 = __builtin_amdgcn_wmma_f32_16x16x32_bf16(false, a2, false, bfrag,
                                                       (short)0, acc2, false, false);
        acc3 = __builtin_amdgcn_wmma_f32_16x16x32_bf16(false, a3, false, bfrag,
                                                       (short)0, acc3, false, false);
    }

    const float bv = bias ? bias[n0 + lmod] : 0.0f;
    const int col = n0 + lmod;
    auto store_tile = [&](const v8f& acc, int rbase) {
#pragma unroll
        for (int j = 0; j < 8; ++j) {
            const float v = acc[j] + bv;            // C layout: VGPR j -> M=j / j+8
            float* p = C + (size_t)(rbase + j + lhalf * 8) * N + col;
            if (NT) __builtin_nontemporal_store(v, p);
            else    *p = v;
        }
    };
    store_tile(acc0, r0);
    store_tile(acc1, r0 + 16);
    store_tile(acc2, r0 + 32);
    store_tile(acc3, r0 + 48);
}

// ---------------------------------------------------------------------------
// Stage W_hh [H,H] into padded LDS (row stride WHH_PAD). 256 threads.
// ---------------------------------------------------------------------------
static __device__ __forceinline__ void stage_whh(float* whh_lds,
                                                 const float* __restrict__ Whh,
                                                 int j) {
    const float* src = Whh + (size_t)j * HH;
    float* dst = whh_lds + (size_t)j * WHH_PAD;
#pragma unroll 8
    for (int k = 0; k < HH; k += 2) {
        dst[k]     = src[k];
        dst[k + 1] = src[k + 1];
    }
}

// ---------------------------------------------------------------------------
// Encoder recurrence: one block per batch element, 256 threads (thread = h idx)
// h_new = tanh(xih[b,t,:] + b_ih + b_hh + h @ W_hh^T); pack semantics.
// Dynamic LDS: W_hh staged once (serial chain then runs on LDS latency).
// ---------------------------------------------------------------------------
__global__ void encoder_rnn_kernel(const float* __restrict__ xih,     // [B*S,H]
                                   const float* __restrict__ Whh,     // [H,H]
                                   const float* __restrict__ b_ih,
                                   const float* __restrict__ b_hh,
                                   const int* __restrict__ lengths,   // [B]
                                   float* __restrict__ enc_out,       // [B,S,H]
                                   float* __restrict__ h_final) {     // [B,H]
    extern __shared__ float whh_lds[];            // HH * WHH_PAD floats
    __shared__ float h[HH];
    const int b = blockIdx.x;
    const int j = threadIdx.x;
    stage_whh(whh_lds, Whh, j);
    h[j] = 0.0f;
    const float bias = b_ih[j] + b_hh[j];
    const int len = lengths[b];
    const float* wrow = whh_lds + (size_t)j * WHH_PAD;
    __syncthreads();
    for (int t = 0; t < SS; ++t) {
        float acc = xih[(size_t)(b * SS + t) * HH + j] + bias;
#pragma unroll 8
        for (int k = 0; k < HH; ++k) acc += h[k] * wrow[k];
        const float hn = tanhf(acc);
        const bool valid = (t < len);
        __syncthreads();
        if (valid) h[j] = hn;
        enc_out[(size_t)(b * SS + t) * HH + j] = valid ? hn : 0.0f;
        __syncthreads();
    }
    h_final[(size_t)b * HH + j] = h[j];
}

// ---------------------------------------------------------------------------
// Decoder recurrence + attention: one block per batch element, 256 threads.
// Emits attn_out rows as bf16 at row (b*T + t) so the logits GEMM writes
// d_out directly in [B,T,V] order.
// ---------------------------------------------------------------------------
__global__ void decoder_rnn_attn_kernel(const float* __restrict__ xih,    // [B*T,H]
                                        const float* __restrict__ Whh,    // [H,H]
                                        const float* __restrict__ b_ih,
                                        const float* __restrict__ b_hh,
                                        const float* __restrict__ Wattn,  // [H,2H]
                                        const float* __restrict__ battn,  // [H]
                                        const float* __restrict__ enc_out,// [B,S,H]
                                        const int* __restrict__ indices_de,
                                        const float* __restrict__ h_init, // [B,H]
                                        bf16_t* __restrict__ Aout) {      // [B*T,H] bf16
    extern __shared__ float whh_lds[];            // HH * WHH_PAD floats
    __shared__ float h[HH];
    __shared__ float comb[2 * HH];
    __shared__ float sc[SS];
    __shared__ float red[SS];
    const int b = blockIdx.x;
    const int j = threadIdx.x;
    stage_whh(whh_lds, Whh, j);
    h[j] = h_init[(size_t)b * HH + j];
    const float bias = b_ih[j] + b_hh[j];
    const float* wrow  = whh_lds + (size_t)j * WHH_PAD;
    const float* warow = Wattn + (size_t)j * (2 * HH);
    const float* eb = enc_out + (size_t)b * SS * HH;
    __syncthreads();
    for (int t = 0; t < TT; ++t) {
        // RNN cell
        float acc = xih[(size_t)(b * TT + t) * HH + j] + bias;
#pragma unroll 8
        for (int k = 0; k < HH; ++k) acc += h[k] * wrow[k];
        const float hn = tanhf(acc);
        __syncthreads();
        h[j] = hn;
        comb[HH + j] = hn;
        __syncthreads();
        // attention scores (threads 0..127, one per source position)
        if (j < SS) {
            float s = 0.0f;
            const float* er = eb + (size_t)j * HH;
#pragma unroll 8
            for (int k = 0; k < HH; ++k) s += h[k] * er[k];
            if (indices_de[b * SS + j] == 0) s = NEGF;
            sc[j] = s; red[j] = s;
        }
        __syncthreads();
        for (int off = 64; off > 0; off >>= 1) {
            if (j < off) red[j] = fmaxf(red[j], red[j + off]);
            __syncthreads();
        }
        const float m = red[0];
        __syncthreads();
        if (j < SS) { const float e = __expf(sc[j] - m); sc[j] = e; red[j] = e; }
        __syncthreads();
        for (int off = 64; off > 0; off >>= 1) {
            if (j < off) red[j] += red[j + off];
            __syncthreads();
        }
        const float inv = 1.0f / red[0];
        __syncthreads();
        // context (thread j = hidden index)
        float ctx = 0.0f;
#pragma unroll 4
        for (int s = 0; s < SS; ++s) ctx += sc[s] * eb[(size_t)s * HH + j];
        comb[j] = ctx * inv;
        __syncthreads();
        // attn_out = tanh([ctx, h] @ W_attn^T + b_attn)
        float a2 = battn[j];
#pragma unroll 8
        for (int c = 0; c < 2 * HH; ++c) a2 += comb[c] * warow[c];
        Aout[(size_t)(b * TT + t) * HH + j] = f2bf(tanhf(a2));
        __syncthreads();
    }
}

// ---------------------------------------------------------------------------
// Host launcher
// ---------------------------------------------------------------------------
extern "C" void kernel_launch(void* const* d_in, const int* in_sizes, int n_in,
                              void* d_out, int out_size, void* d_ws, size_t ws_size,
                              hipStream_t stream) {
    const int*   indices_de = (const int*)d_in[0];
    const int*   lengths_de = (const int*)d_in[1];
    const int*   indices_en = (const int*)d_in[2];
    /* lengths_en (d_in[3]) unused: always full length */
    const float* emb_enc  = (const float*)d_in[4];
    const float* emb_dec  = (const float*)d_in[5];
    const float* W_ih_enc = (const float*)d_in[6];
    const float* W_hh_enc = (const float*)d_in[7];
    const float* b_ih_enc = (const float*)d_in[8];
    const float* b_hh_enc = (const float*)d_in[9];
    const float* W_ih_dec = (const float*)d_in[10];
    const float* W_hh_dec = (const float*)d_in[11];
    const float* b_ih_dec = (const float*)d_in[12];
    const float* b_hh_dec = (const float*)d_in[13];
    const float* W_attn   = (const float*)d_in[14];
    const float* b_attn   = (const float*)d_in[15];
    const float* W_out    = (const float*)d_in[16];
    const float* b_out    = (const float*)d_in[17];
    float* out = (float*)d_out;

    // Workspace carve-out (all offsets 256B aligned)
    char* ws = (char*)d_ws;
    size_t off = 0;
    auto carve = [&](size_t bytes) -> void* {
        void* p = ws + off;
        off += (bytes + 255) & ~(size_t)255;
        return p;
    };
    bf16_t* WihE_b = (bf16_t*)carve((size_t)HH * HH * 2);        // 128 KB
    bf16_t* WihD_b = (bf16_t*)carve((size_t)HH * HH * 2);        // 128 KB
    bf16_t* Wout_b = (bf16_t*)carve((size_t)VV * HH * 2);        // 16.4 MB
    bf16_t* Xenc_b = (bf16_t*)carve((size_t)BB * SS * HH * 2);   // 2 MB
    bf16_t* Xdec_b = (bf16_t*)carve((size_t)BB * TT * HH * 2);   // 2 MB
    float*  xihE   = (float*)carve((size_t)BB * SS * HH * 4);    // 4 MB
    float*  xihD   = (float*)carve((size_t)BB * TT * HH * 4);    // 4 MB
    float*  encO   = (float*)carve((size_t)BB * SS * HH * 4);    // 4 MB
    float*  hFin   = (float*)carve((size_t)BB * HH * 4);         // 32 KB
    bf16_t* Alog   = (bf16_t*)carve((size_t)BB * TT * HH * 2);   // 2 MB
    (void)ws_size; (void)in_sizes; (void)n_in; (void)out_size;

    // 1. weight conversions
    f32_to_bf16_kernel<<<128, 256, 0, stream>>>(W_ih_enc, WihE_b, (size_t)HH * HH);
    f32_to_bf16_kernel<<<128, 256, 0, stream>>>(W_ih_dec, WihD_b, (size_t)HH * HH);
    f32_to_bf16_kernel<<<2048, 256, 0, stream>>>(W_out, Wout_b, (size_t)VV * HH);

    // 2. embedding gathers
    gather_embed_kernel<<<BB * SS, HH, 0, stream>>>(emb_enc, indices_de, Xenc_b);
    gather_embed_kernel<<<BB * TT, HH, 0, stream>>>(emb_dec, indices_en, Xdec_b);

    // 3. hoisted input projections (WMMA):  xih = X @ W_ih^T  (RT stores: re-read)
    {
        const int M = BB * SS, N = HH, K = HH;
        const int groups = (M / 64) * (N / 16);         // 1024
        gemm_bf16_wmma_kernel<false><<<(groups + 7) / 8, 256, 0, stream>>>(
            Xenc_b, WihE_b, nullptr, xihE, M, N, K, groups);
        gemm_bf16_wmma_kernel<false><<<(groups + 7) / 8, 256, 0, stream>>>(
            Xdec_b, WihD_b, nullptr, xihD, M, N, K, groups);
    }

    // 4. encoder recurrence (batch-parallel, W_hh in LDS)
    encoder_rnn_kernel<<<BB, HH, WHH_LDS_BYTES, stream>>>(
        xihE, W_hh_enc, b_ih_enc, b_hh_enc, lengths_de, encO, hFin);

    // 5. decoder recurrence + attention (batch-parallel, W_hh in LDS)
    decoder_rnn_attn_kernel<<<BB, HH, WHH_LDS_BYTES, stream>>>(
        xihD, W_hh_dec, b_ih_dec, b_hh_dec, W_attn, b_attn, encO, indices_de,
        hFin, Alog);

    // 6. vocab projection (dominant; NT stores keep W_out L2-resident,
    //    output is pure 524 MB stream -> ~22.5 us floor at 23.3 TB/s)
    {
        const int M = BB * TT, N = VV, K = HH;
        const int groups = (M / 64) * (N / 16);         // 128000
        gemm_bf16_wmma_kernel<true><<<(groups + 7) / 8, 256, 0, stream>>>(
            Alog, Wout_b, b_out, out, M, N, K, groups);
    }
}
